// LSTM_34059090657651
// MI455X (gfx1250) — matrix operand order repaired and checked by reading, not compiled, not run on hardware
//
#include <hip/hip_runtime.h>
#include <stdint.h>

// ---------------- problem constants ----------------
#define Bn 32
#define Tn 2048
#define Dn 256
#define Hn 512
#define On 256
#define G4H (4 * Hn)      // 2048

typedef unsigned short u16;
typedef __attribute__((ext_vector_type(16))) __bf16 v16bf;
typedef __attribute__((ext_vector_type(8)))  float  v8f;

union Frag32 {            // 16 bf16 = 32 bytes = two uint4 chunks
    uint4 q[2];
    v16bf v;
};

__device__ __forceinline__ u16 f2bf(float f) {
    union { float f; unsigned u; } v; v.f = f;
    unsigned u = v.u + 0x7FFFu + ((v.u >> 16) & 1u);   // RNE
    return (u16)(u >> 16);
}
__device__ __forceinline__ float bf2f(u16 h) {
    union { unsigned u; float f; } v; v.u = ((unsigned)h) << 16;
    return v.f;
}
__device__ __forceinline__ float sigmf(float x) {
    return 1.0f / (1.0f + __expf(-x));
}
// saturating tanh via exp(-2|x|): no NaN at large |x|, pure VALU+trans
__device__ __forceinline__ float tanh_fast(float x) {
    float e = __expf(-2.0f * fabsf(x));
    float t = (1.0f - e) / (1.0f + e);
    return copysignf(t, x);
}

// ---------------- pack / init kernels ----------------
__global__ void pack_x_kernel(const float* __restrict__ x, u16* __restrict__ xbf) {
    size_t idx = (size_t)blockIdx.x * blockDim.x + threadIdx.x;
    if (idx >= (size_t)Tn * Bn * Dn) return;
    size_t t = idx / (Bn * Dn);
    size_t r = idx % (Bn * Dn);
    size_t b = r / Dn, d = r % Dn;
    xbf[idx] = f2bf(x[(b * Tn + t) * Dn + d]);   // [B,T,D] -> [T,B,D] bf16
}

__global__ void f32_to_bf16_kernel(const float* __restrict__ src, u16* __restrict__ dst, int n) {
    int i = blockIdx.x * blockDim.x + threadIdx.x;
    if (i < n) dst[i] = f2bf(src[i]);
}

__global__ void bias_init_kernel(const float* bi0, const float* bh0,
                                 const float* bi1, const float* bh1,
                                 float* bias0, float* bias1) {
    int i = blockIdx.x * blockDim.x + threadIdx.x;
    if (i < G4H) {
        bias0[i] = bi0[i] + bh0[i];
        bias1[i] = bi1[i] + bh1[i];
    }
}

__global__ void zero_init_kernel(u16* hb0, u16* hb1, unsigned* bars) {
    int i = blockIdx.x * blockDim.x + threadIdx.x;
    if (i < 2 * Bn * Hn) { hb0[i] = 0; hb1[i] = 0; }
    if (i < 8) bars[i] = 0;
}

// ---------------- persistent fused LSTM layer ----------------
// Grid: 32 blocks x 256 threads. Block g owns hidden units [g*16, g*16+16).
// LDS: W slice [64][Din+H] bf16 (W_ih || W_hh rows), then gate buffer [64][32] f32.
__global__ __launch_bounds__(256) void lstm_layer_kernel(
    const u16* __restrict__ xin,   // [T][B][Din] bf16
    int Din,
    const u16* __restrict__ Wih,   // [4H][Din] bf16
    const u16* __restrict__ Whh,   // [4H][H]  bf16
    const float* __restrict__ bias,// [4H] = b_ih + b_hh
    u16* __restrict__ hbuf,        // [2][B][H] bf16 ping-pong, buffer 0 pre-zeroed
    u16* __restrict__ hs_out,      // [T][B][H] bf16 or nullptr
    unsigned* __restrict__ bar)    // monotonic grid-barrier counter (pre-zeroed)
{
    extern __shared__ u16 smem[];
    const int K = Din + Hn;                 // fused K dimension
    u16*   Wlds = smem;                     // [64][K]
    float* gbuf = (float*)(smem + 64 * K);  // [64 cols][32 batch]

    const int g   = blockIdx.x;
    const int j0  = g * 16;                 // first hidden unit of this block
    const int tid = threadIdx.x;

    // ---- stage weight slice into LDS once via async DMA (no VGPR bounce) ----
    // 64*(K/8) 16-byte chunks; exact multiple of 256 threads for both layers.
    for (int c = tid; c < 64 * (K / 8); c += 256) {
        int row = c / (K / 8);
        int k   = (c % (K / 8)) * 8;
        int q   = row >> 4, jj = row & 15;
        int n_g = q * Hn + j0 + jj;         // global gate column
        const u16* src = (k < Din) ? (Wih + (size_t)n_g * Din + k)
                                   : (Whh + (size_t)n_g * Hn + (k - Din));
        unsigned dst = (unsigned)(size_t)(Wlds + row * K + k);  // LDS byte offset
        asm volatile("global_load_async_to_lds_b128 %0, %1, off"
                     :: "v"(dst), "v"(src) : "memory");
    }
    asm volatile("s_wait_asynccnt 0x0" ::: "memory");
    __syncthreads();

    // ---- wave / tile assignment ----
    const int wave  = tid >> 5;             // 0..7
    const int lane  = tid & 31;
    const int l     = lane & 15;
    const int half  = lane >> 4;
    const int mtile = wave & 1;             // batch tile (B=32 -> 2 tiles)
    const int qgate = wave >> 1;            // gate 0..3 == local N-tile
    const int m0    = mtile * 16;
    const int n0l   = qgate * 16;           // local column base

    const u16* brow = Wlds + (n0l + l) * K + half * 16;  // B-frag row (LDS)
    const float bb  = bias[qgate * Hn + j0 + l];

    float creg[2] = {0.0f, 0.0f};           // cell state lives in registers

    for (int t = 0; t < Tn; ++t) {
        const u16* hprev = hbuf + (size_t)(t & 1) * (Bn * Hn);
        u16*       hnext = hbuf + (size_t)((t + 1) & 1) * (Bn * Hn);
        const u16* xt    = xin + (size_t)t * Bn * Din;

        // prefetch next step's x rows (only HBM stream on the critical path)
        if (t + 1 < Tn) {
            const char* nx = (const char*)(xin + (size_t)(t + 1) * Bn * Din
                                               + (m0 + l) * Din);
            for (int pb = 0; pb < Din * 2; pb += 256)
                __builtin_prefetch(nx + pb, 0, 1);
        }

        // dual accumulators break the WMMA D->C RAW chain (XDL ILP)
        v8f acc0 = {0.f, 0.f, 0.f, 0.f, 0.f, 0.f, 0.f, 0.f};
        v8f acc1 = {0.f, 0.f, 0.f, 0.f, 0.f, 0.f, 0.f, 0.f};

        // ---- x_t contribution (K = 0..Din) ----
        const u16* arow_x = xt + (m0 + l) * Din;
        #pragma unroll 2
        for (int k0 = 0; k0 < Din; k0 += 64) {
            Frag32 a0, b0, a1, b1;
            a0.q[0] = *reinterpret_cast<const uint4*>(arow_x + k0 + half * 16);
            a0.q[1] = *reinterpret_cast<const uint4*>(arow_x + k0 + 8 + half * 16);
            b0.q[0] = *reinterpret_cast<const uint4*>(brow + k0);
            b0.q[1] = *reinterpret_cast<const uint4*>(brow + k0 + 8);
            a1.q[0] = *reinterpret_cast<const uint4*>(arow_x + k0 + 32 + half * 16);
            a1.q[1] = *reinterpret_cast<const uint4*>(arow_x + k0 + 40 + half * 16);
            b1.q[0] = *reinterpret_cast<const uint4*>(brow + k0 + 32);
            b1.q[1] = *reinterpret_cast<const uint4*>(brow + k0 + 40);
            acc0 = __builtin_amdgcn_wmma_f32_16x16x32_bf16(
                false, a0.v, false, b0.v, (short)0, acc0, false, false);
            acc1 = __builtin_amdgcn_wmma_f32_16x16x32_bf16(
                false, a1.v, false, b1.v, (short)0, acc1, false, false);
        }

        // ---- h_{t-1} contribution (K = Din..Din+H) ----
        const u16* arow_h = hprev + (m0 + l) * Hn;
        const u16* browh  = brow + Din;
        #pragma unroll 2
        for (int k0 = 0; k0 < Hn; k0 += 64) {
            Frag32 a0, b0, a1, b1;
            a0.q[0] = *reinterpret_cast<const uint4*>(arow_h + k0 + half * 16);
            a0.q[1] = *reinterpret_cast<const uint4*>(arow_h + k0 + 8 + half * 16);
            b0.q[0] = *reinterpret_cast<const uint4*>(browh + k0);
            b0.q[1] = *reinterpret_cast<const uint4*>(browh + k0 + 8);
            a1.q[0] = *reinterpret_cast<const uint4*>(arow_h + k0 + 32 + half * 16);
            a1.q[1] = *reinterpret_cast<const uint4*>(arow_h + k0 + 40 + half * 16);
            b1.q[0] = *reinterpret_cast<const uint4*>(browh + k0 + 32);
            b1.q[1] = *reinterpret_cast<const uint4*>(browh + k0 + 40);
            acc0 = __builtin_amdgcn_wmma_f32_16x16x32_bf16(
                false, a0.v, false, b0.v, (short)0, acc0, false, false);
            acc1 = __builtin_amdgcn_wmma_f32_16x16x32_bf16(
                false, a1.v, false, b1.v, (short)0, acc1, false, false);
        }
        v8f acc = acc0 + acc1;

        // ---- bias + scatter C tile to LDS gate buffer ----
        #pragma unroll
        for (int v = 0; v < 8; ++v)
            gbuf[(n0l + l) * 32 + (m0 + half * 8 + v)] = acc[v] + bb;
        __syncthreads();

        // ---- elementwise recurrence: 16 j x 32 b = 512 pairs, 2 per thread ----
        #pragma unroll
        for (int r = 0; r < 2; ++r) {
            int p = tid + r * 256;
            int j = p >> 5, b = p & 31;
            float gi = gbuf[(0 * 16 + j) * 32 + b];
            float gf = gbuf[(1 * 16 + j) * 32 + b];
            float gg = gbuf[(2 * 16 + j) * 32 + b];
            float go = gbuf[(3 * 16 + j) * 32 + b];
            float c  = sigmf(gf) * creg[r] + sigmf(gi) * tanh_fast(gg);
            creg[r]  = c;
            float h  = sigmf(go) * tanh_fast(c);
            u16 hb   = f2bf(h);
            hnext[b * Hn + j0 + j] = hb;
            if (hs_out) hs_out[(size_t)t * Bn * Hn + b * Hn + j0 + j] = hb;
        }

        // ---- grid barrier (monotonic counter, 32 co-resident WGs) ----
        __threadfence();
        __syncthreads();
        if (tid == 0) {
            __hip_atomic_fetch_add(bar, 1u, __ATOMIC_RELEASE,
                                   __HIP_MEMORY_SCOPE_AGENT);
            unsigned target = (unsigned)(t + 1) * gridDim.x;
            while (__hip_atomic_load(bar, __ATOMIC_ACQUIRE,
                                     __HIP_MEMORY_SCOPE_AGENT) < target) {
                __builtin_amdgcn_s_sleep(1);
            }
        }
        __syncthreads();
        __threadfence();
    }
}

// ---------------- final linear: out = h1_last @ W_out^T + b_out ----------------
__global__ void out_linear_kernel(const u16* __restrict__ h1,     // [B][H] bf16
                                  const float* __restrict__ W_out,// [O][H]
                                  const float* __restrict__ b_out,// [O]
                                  float* __restrict__ out)        // [B][O]
{
    int idx = blockIdx.x * blockDim.x + threadIdx.x;
    if (idx >= Bn * On) return;
    int b = idx / On, o = idx % On;
    float s = b_out[o];
    const u16*   hr = h1 + b * Hn;
    const float* wr = W_out + (size_t)o * Hn;
    for (int k = 0; k < Hn; ++k) s += bf2f(hr[k]) * wr[k];
    out[idx] = s;
}

// ---------------- launch ----------------
extern "C" void kernel_launch(void* const* d_in, const int* in_sizes, int n_in,
                              void* d_out, int out_size, void* d_ws, size_t ws_size,
                              hipStream_t stream) {
    (void)in_sizes; (void)n_in; (void)out_size; (void)ws_size;

    const float* x     = (const float*)d_in[0];
    const float* W_ih0 = (const float*)d_in[1];
    const float* W_hh0 = (const float*)d_in[2];
    const float* b_ih0 = (const float*)d_in[3];
    const float* b_hh0 = (const float*)d_in[4];
    const float* W_ih1 = (const float*)d_in[5];
    const float* W_hh1 = (const float*)d_in[6];
    const float* b_ih1 = (const float*)d_in[7];
    const float* b_hh1 = (const float*)d_in[8];
    const float* W_out = (const float*)d_in[9];
    const float* b_out = (const float*)d_in[10];
    float* out = (float*)d_out;

    // workspace carve-up (256B aligned regions)
    char* ws = (char*)d_ws;
    size_t off = 0;
    auto carve = [&](size_t bytes) -> char* {
        char* p = ws + off;
        off = (off + bytes + 255) & ~(size_t)255;
        return p;
    };
    unsigned* bars  = (unsigned*)carve(64 * sizeof(unsigned));
    u16* xbf   = (u16*)carve((size_t)Tn * Bn * Dn * 2);
    u16* wih0  = (u16*)carve((size_t)G4H * Dn * 2);
    u16* whh0  = (u16*)carve((size_t)G4H * Hn * 2);
    u16* wih1  = (u16*)carve((size_t)G4H * Hn * 2);
    u16* whh1  = (u16*)carve((size_t)G4H * Hn * 2);
    float* bias0 = (float*)carve(G4H * sizeof(float));
    float* bias1 = (float*)carve(G4H * sizeof(float));
    u16* hbuf0 = (u16*)carve((size_t)2 * Bn * Hn * 2);
    u16* hbuf1 = (u16*)carve((size_t)2 * Bn * Hn * 2);
    u16* hs0   = (u16*)carve((size_t)Tn * Bn * Hn * 2);

    // init (re-run every call so graph replay is deterministic)
    zero_init_kernel<<<(2 * Bn * Hn + 255) / 256, 256, 0, stream>>>(hbuf0, hbuf1, bars);

    // pack to bf16
    {
        size_t n = (size_t)Tn * Bn * Dn;
        pack_x_kernel<<<(unsigned)((n + 255) / 256), 256, 0, stream>>>(x, xbf);
    }
    f32_to_bf16_kernel<<<(G4H * Dn + 255) / 256, 256, 0, stream>>>(W_ih0, wih0, G4H * Dn);
    f32_to_bf16_kernel<<<(G4H * Hn + 255) / 256, 256, 0, stream>>>(W_hh0, whh0, G4H * Hn);
    f32_to_bf16_kernel<<<(G4H * Hn + 255) / 256, 256, 0, stream>>>(W_ih1, wih1, G4H * Hn);
    f32_to_bf16_kernel<<<(G4H * Hn + 255) / 256, 256, 0, stream>>>(W_hh1, whh1, G4H * Hn);
    bias_init_kernel<<<(G4H + 255) / 256, 256, 0, stream>>>(b_ih0, b_hh0, b_ih1, b_hh1, bias0, bias1);

    // layer 0: Din = D, writes hs0 (time-major) for layer 1
    {
        int K = Dn + Hn;                                   // 768
        size_t shmem = (size_t)64 * K * 2 + 64 * 32 * 4;   // 106496 B
        lstm_layer_kernel<<<32, 256, shmem, stream>>>(
            xbf, Dn, wih0, whh0, bias0, hbuf0, hs0, bars + 0);
    }
    // layer 1: Din = H, input = hs0, no sequence output needed
    {
        int K = Hn + Hn;                                   // 1024
        size_t shmem = (size_t)64 * K * 2 + 64 * 32 * 4;   // 139264 B
        lstm_layer_kernel<<<32, 256, shmem, stream>>>(
            hs0, Hn, wih1, whh1, bias1, hbuf1, (u16*)nullptr, bars + 16);
    }

    // final h of layer 1 landed in hbuf1 buffer (T & 1)==0
    out_linear_kernel<<<(Bn * On + 255) / 256, 256, 0, stream>>>(
        hbuf1, W_out, b_out, out);
}
